// NewsEncoder_84911503442401
// MI455X (gfx1250) — compile-verified, matrix operand order checked by reading, not run on hardware
//
#include <hip/hip_runtime.h>

typedef unsigned short ushort_t;
typedef __attribute__((ext_vector_type(16))) __bf16 v16bf;
typedef __attribute__((ext_vector_type(8)))  float  v8f;

#define GAS __attribute__((address_space(1)))

#define D_   150
#define H_   64
#define NL_  4
#define B_   8
#define S_   64
#define NT_  6
#define LT_  32
#define NC_  6
#define LC_  128
#define N_   512     // B*S
#define G_   192     // 3*H
#define KP0_ 160     // D padded to multiple of 32

// ---------- helpers ----------

__device__ __forceinline__ ushort_t f2bf(float f) {
  union { float f; unsigned u; } x; x.f = f;
  unsigned u = x.u;
  unsigned r = u + 0x7FFFu + ((u >> 16) & 1u);   // round-to-nearest-even
  return (ushort_t)(r >> 16);
}

__device__ __forceinline__ float fast_sigmoid(float x) {
  // v_exp + v_rcp (TRANS ops), avoids IEEE division expansion
  return __builtin_amdgcn_rcpf(1.f + __expf(-x));
}
__device__ __forceinline__ float fast_tanh(float x) {
  // tanh(x) = 1 - 2/(exp(2x)+1)
  return 1.f - 2.f * __builtin_amdgcn_rcpf(1.f + __expf(2.f * x));
}

// Force global (addrspace 1) access — keeps loads/stores off the FLAT path so
// they use only LOADcnt/STOREcnt and never couple with DScnt.
__device__ __forceinline__ const GAS float* gcf(const float* p) {
  return (const GAS float*)(unsigned long long)p;
}
__device__ __forceinline__ GAS float* gf(float* p) {
  return (GAS float*)(unsigned long long)p;
}
__device__ __forceinline__ GAS ushort_t* gus(ushort_t* p) {
  return (GAS ushort_t*)(unsigned long long)p;
}

union Frag16 { v16bf v; uint4 q[2]; ushort_t h[16]; };

// Load a 16x32 bf16 fragment (A-layout; also valid as B when base = W + g0*stride,
// since W is stored [192,K] row-major == B^T row-major).
__device__ __forceinline__ v16bf load_frag_g(const ushort_t* base, int stride, int k0) {
  int lane = threadIdx.x & 31;
  const ushort_t* p = base + (long)(lane & 15) * stride + k0 + ((lane >> 4) << 3);
  Frag16 f;
  f.q[0] = *(const uint4*)(p);        // K = k0+ksub .. +7
  f.q[1] = *(const uint4*)(p + 16);   // K = k0+ksub+16 .. +23
  return f.v;
}

#define WMMA_BF16(a, b, c) \
  __builtin_amdgcn_wmma_f32_16x16x32_bf16(false, (a), false, (b), (short)0, (c), false, false)

// ---------- kernels ----------

// Embedding-bag: sum emb rows over NJ news per (n,t), write bf16 padded row [KP0_]
__global__ void embed_sum_kernel(const int* __restrict__ tok, const float* __restrict__ emb,
                                 ushort_t* __restrict__ xb, int T, int NJ) {
  int blk = blockIdx.x;
  int t = blk % T;
  long n = blk / T;
  int d = threadIdx.x;                 // 0..159
  float s = 0.f;
  if (d < D_) {
    const int* tp = tok + (n * NJ) * T + t;   // tok[n, j, t]
    for (int j = 0; j < NJ; ++j) {
      int ix = tp[j * T];
      if (ix > 0) s += emb[(long)ix * D_ + d];   // padding_idx=0 contributes 0
    }
  }
  xb[(n * T + t) * KP0_ + d] = f2bf(s);
}

// xg[m,g] = bias[g] + sum_k xb[m,k]*W[g,k]  (N=192, K=NK*32).
// B fragments (weights) preloaded into VGPRs once; MT M-tiles per block amortize them.
template <int NK, int MT>
__global__ void __launch_bounds__(128, 1)
gemm_xg_kernel(const ushort_t* __restrict__ xb, const ushort_t* __restrict__ W,
               const float* __restrict__ bias, float* __restrict__ xg) {
  constexpr int Kp = NK * 32;
  int wv   = (threadIdx.x >> 5) & 3;
  int lane = threadIdx.x & 31;
  int col  = lane & 15, hi = lane >> 4;
  long mbase = (long)blockIdx.x * (16 * MT);

  // Preload all B fragments: 3 gate tiles x NK k-chunks, resident in VGPRs.
  v16bf wB[3][NK];
#pragma unroll
  for (int g = 0; g < 3; ++g)
#pragma unroll
    for (int c = 0; c < NK; ++c)
      wB[g][c] = load_frag_g(W + (long)(64 * g + 16 * wv) * Kp, Kp, 32 * c);

  float b0 = bias[16 * wv + col];
  float b1 = bias[64 + 16 * wv + col];
  float b2 = bias[128 + 16 * wv + col];

#pragma unroll
  for (int it = 0; it < MT; ++it) {
    const ushort_t* arow = xb + (mbase + it * 16) * Kp;
    v16bf a[NK];
#pragma unroll
    for (int c = 0; c < NK; ++c) a[c] = load_frag_g(arow, Kp, 32 * c);

    v8f acc0, acc1, acc2;
#pragma unroll
    for (int i = 0; i < 8; ++i) { acc0[i] = b0; acc1[i] = b1; acc2[i] = b2; }
#pragma unroll
    for (int c = 0; c < NK; ++c) {
      acc0 = WMMA_BF16(a[c], wB[0][c], acc0);
      acc1 = WMMA_BF16(a[c], wB[1][c], acc1);
      acc2 = WMMA_BF16(a[c], wB[2][c], acc2);
    }
#pragma unroll
    for (int i = 0; i < 8; ++i) {
      long m = mbase + it * 16 + hi * 8 + i;
      GAS float* row = gf(xg + m * (long)G_);
      row[16 * wv + col]       = acc0[i];
      row[64 + 16 * wv + col]  = acc1[i];
      row[128 + 16 * wv + col] = acc2[i];
    }
  }
}

// GRU recurrence: 16 sequences per block, 4 waves; wave w owns h-columns [16w,16w+16).
// Per step: gh = h @ Whh^T + bhh (6 bf16 WMMAs), gate math in f32 with fast trans-op
// sigmoid/tanh, h kept in v8f regs, bf16 copy exchanged through LDS for the next A.
// All global addresses are incremented pointers; accesses forced to addrspace(1).
__global__ void __launch_bounds__(128, 1)
gru_rec_kernel(const float* __restrict__ xg, const ushort_t* __restrict__ Whh,
               const float* __restrict__ bhh, int T,
               ushort_t* __restrict__ xb_out, int outT, int t_off,
               float* __restrict__ y_out, float* __restrict__ hid_out, int layer) {
  __shared__ __align__(16) ushort_t hbf[16 * 64];
  int wv   = (threadIdx.x >> 5) & 3;
  int lane = threadIdx.x & 31;
  int col  = lane & 15, hi = lane >> 4;
  long nbase = (long)blockIdx.x * 16;
  int j = 16 * wv + col;               // this lane's h column

  v16bf br0 = load_frag_g(Whh + (long)(16 * wv) * 64,        64, 0);
  v16bf br1 = load_frag_g(Whh + (long)(16 * wv) * 64,        64, 32);
  v16bf bz0 = load_frag_g(Whh + (long)(64 + 16 * wv) * 64,   64, 0);
  v16bf bz1 = load_frag_g(Whh + (long)(64 + 16 * wv) * 64,   64, 32);
  v16bf bn0 = load_frag_g(Whh + (long)(128 + 16 * wv) * 64,  64, 0);
  v16bf bn1 = load_frag_g(Whh + (long)(128 + 16 * wv) * 64,  64, 32);
  float biasr = bhh[j], biasz = bhh[64 + j], biasn = bhh[128 + j];

  // Per-element xg pointers (global AS); advance by one row (192 floats) per step.
  const GAS float* xp[8];
#pragma unroll
  for (int i = 0; i < 8; ++i)
    xp[i] = gcf(xg + ((nbase + hi * 8 + i) * (long)T) * (long)G_ + j);

  // Output bases advance by 64 per step; per-element offsets are wave-uniform.
  GAS ushort_t* xb_base = gus(xb_out
      ? xb_out + ((nbase + hi * 8) * (long)outT + t_off) * 64L + j : (ushort_t*)0);
  GAS float* y_base = gf(y_out
      ? y_out + ((nbase + hi * 8) * (long)T) * 64L + j : (float*)0);
  int xb_eo = outT * 64;               // element-to-element stride (uniform)
  int y_eo  = T * 64;

  for (int i = threadIdx.x; i < 16 * 64; i += blockDim.x) hbf[i] = 0;
  v8f h;
#pragma unroll
  for (int i = 0; i < 8; ++i) h[i] = 0.f;

  for (int t = 0; t < T; ++t) {
    __syncthreads();                   // hbf (h_t as bf16) visible
    Frag16 fa0, fa1;
    {
      const ushort_t* pr = &hbf[(lane & 15) * 64 + (hi << 3)];
      fa0.q[0] = *(const uint4*)(pr);
      fa0.q[1] = *(const uint4*)(pr + 16);
      fa1.q[0] = *(const uint4*)(pr + 32);
      fa1.q[1] = *(const uint4*)(pr + 48);
    }
    // Prefetch next step's gate inputs (fire-and-forget, no counters).
    if (t + 1 < T) {
#pragma unroll
      for (int i = 0; i < 8; ++i) {
        __builtin_prefetch((const float*)(unsigned long long)(xp[i] + 192), 0, 3);
        __builtin_prefetch((const float*)(unsigned long long)(xp[i] + 192 + 64), 0, 3);
        __builtin_prefetch((const float*)(unsigned long long)(xp[i] + 192 + 128), 0, 3);
      }
    }
    v8f gr, gz, gn;
#pragma unroll
    for (int i = 0; i < 8; ++i) { gr[i] = biasr; gz[i] = biasz; gn[i] = biasn; }
    gr = WMMA_BF16(fa0.v, br0, gr);  gr = WMMA_BF16(fa1.v, br1, gr);
    gz = WMMA_BF16(fa0.v, bz0, gz);  gz = WMMA_BF16(fa1.v, bz1, gz);
    gn = WMMA_BF16(fa0.v, bn0, gn);  gn = WMMA_BF16(fa1.v, bn1, gn);

    // Load all gate inputs first (clauseable global_load_b32), then do the math.
    float xr[8], xz[8], xn[8];
#pragma unroll
    for (int i = 0; i < 8; ++i) {
      xr[i] = xp[i][0];
      xz[i] = xp[i][64];
      xn[i] = xp[i][128];
      xp[i] += G_;
    }
    v8f hnew;
#pragma unroll
    for (int i = 0; i < 8; ++i) {
      float r  = fast_sigmoid(xr[i] + gr[i]);
      float z  = fast_sigmoid(xz[i] + gz[i]);
      float nn = fast_tanh(xn[i] + r * gn[i]);
      hnew[i] = (1.f - z) * nn + z * h[i];
    }
    __syncthreads();                   // all waves done reading old hbf
#pragma unroll
    for (int i = 0; i < 8; ++i) {
      int mloc = hi * 8 + i;
      ushort_t hb = f2bf(hnew[i]);
      hbf[mloc * 64 + j] = hb;
      if (xb_out) xb_base[i * xb_eo] = hb;
      if (y_out)  y_base[i * y_eo] = hnew[i];
      h[i] = hnew[i];
    }
    if (xb_out) xb_base += 64;
    if (y_out)  y_base += 64;
  }
  if (hid_out) {                       // last hidden -> [B, NL, S, H]
#pragma unroll
    for (int i = 0; i < 8; ++i) {
      long n = nbase + hi * 8 + i;
      long b = n >> 6, s = n & 63;
      gf(hid_out)[((b * NL_ + layer) * S_ + s) * (long)H_ + j] = h[i];
    }
  }
}

__global__ void cvt_kernel(const float* __restrict__ src, ushort_t* __restrict__ dst, int n) {
  int i = blockIdx.x * blockDim.x + threadIdx.x;
  if (i < n) dst[i] = f2bf(src[i]);
}

__global__ void cvt_pad_wih0_kernel(const float* __restrict__ src, ushort_t* __restrict__ dst) {
  int r = blockIdx.x, c = threadIdx.x;          // 192 x 160
  float v = (c < D_) ? src[r * D_ + c] : 0.f;
  dst[r * KP0_ + c] = f2bf(v);
}

__global__ void copy_f32_kernel(const float* __restrict__ src, float* __restrict__ dst, int n) {
  int i = blockIdx.x * blockDim.x + threadIdx.x;
  if (i < n) dst[i] = src[i];
}

// ---------- launch ----------

extern "C" void kernel_launch(void* const* d_in, const int* in_sizes, int n_in,
                              void* d_out, int out_size, void* d_ws, size_t ws_size,
                              hipStream_t stream) {
  (void)in_sizes; (void)n_in; (void)out_size; (void)ws_size;
  const int*   titles   = (const int*)d_in[0];
  const int*   contents = (const int*)d_in[1];
  const float* prices   = (const float*)d_in[2];
  const float* emb      = (const float*)d_in[3];
  const float* iWih0    = (const float*)d_in[4];
  const float* iWhh0    = (const float*)d_in[5];
  const float* ibih0    = (const float*)d_in[6];
  const float* ibhh0    = (const float*)d_in[7];
  const float* iWih     = (const float*)d_in[8];
  const float* iWhh     = (const float*)d_in[9];
  const float* ibih     = (const float*)d_in[10];
  const float* ibhh     = (const float*)d_in[11];
  const float* cWih     = (const float*)d_in[12];
  const float* cWhh     = (const float*)d_in[13];
  const float* cbih     = (const float*)d_in[14];
  const float* cbhh     = (const float*)d_in[15];

  char* ws = (char*)d_ws;
  size_t off = 0;
  auto alloc = [&](size_t bytes) -> void* {
    void* p = ws + off;
    off = (off + bytes + 255) & ~(size_t)255;
    return p;
  };
  ushort_t* xb160 = (ushort_t*)alloc((size_t)N_ * 128 * 160 * 2); // layer-0 bf16 input
  ushort_t* xb64  = (ushort_t*)alloc((size_t)N_ * 160 * 64 * 2);  // layers>=1 bf16 input
  ushort_t* catb  = (ushort_t*)alloc((size_t)N_ * 160 * 64 * 2);  // [ot|oc] comb input
  float*    xg    = (float*)alloc((size_t)N_ * 160 * 192 * 4);    // input projection
  ushort_t* wih0b = (ushort_t*)alloc((size_t)192 * 160 * 2);
  ushort_t* wblk  = (ushort_t*)alloc((size_t)15 * 192 * 64 * 2);
  const int WSZ = 192 * 64;
  ushort_t* iWihB = wblk;             // 3: info Wih layers 1..3
  ushort_t* iWhhB = wblk + 3 * WSZ;   // 4: info Whh layers 0..3
  ushort_t* cWihB = wblk + 7 * WSZ;   // 4: comb Wih
  ushort_t* cWhhB = wblk + 11 * WSZ;  // 4: comb Whh

  float* outE = (float*)d_out;                 // [512,160,64]
  float* outH = (float*)d_out + 5242880;       // [B,NL,S,H]
  float* outP = (float*)d_out + 5242880 + 131072;

  // weight prep (bf16) + prices passthrough
  cvt_pad_wih0_kernel<<<192, 160, 0, stream>>>(iWih0, wih0b);
  cvt_kernel<<<(3 * WSZ + 255) / 256, 256, 0, stream>>>(iWih, iWihB, 3 * WSZ);
  cvt_kernel<<<(WSZ + 255) / 256, 256, 0, stream>>>(iWhh0, iWhhB, WSZ);
  cvt_kernel<<<(3 * WSZ + 255) / 256, 256, 0, stream>>>(iWhh, iWhhB + WSZ, 3 * WSZ);
  cvt_kernel<<<(4 * WSZ + 255) / 256, 256, 0, stream>>>(cWih, cWihB, 4 * WSZ);
  cvt_kernel<<<(4 * WSZ + 255) / 256, 256, 0, stream>>>(cWhh, cWhhB, 4 * WSZ);
  copy_f32_kernel<<<2, 256, 0, stream>>>(prices, outP, 512);

  // -------- titles path (T = 32, M = 16384) --------
  embed_sum_kernel<<<N_ * LT_, 160, 0, stream>>>(titles, emb, xb160, LT_, NT_);
  gemm_xg_kernel<5, 2><<<N_ * LT_ / 32, 128, 0, stream>>>(xb160, wih0b, ibih0, xg);
  gru_rec_kernel<<<N_ / 16, 128, 0, stream>>>(xg, iWhhB, ibhh0, LT_, xb64, LT_, 0,
                                              nullptr, nullptr, 0);
  for (int l = 1; l < 4; ++l) {
    gemm_xg_kernel<2, 4><<<N_ * LT_ / 64, 128, 0, stream>>>(xb64, iWihB + (l - 1) * WSZ,
                                                            ibih + (l - 1) * G_, xg);
    bool last = (l == 3);
    gru_rec_kernel<<<N_ / 16, 128, 0, stream>>>(xg, iWhhB + l * WSZ, ibhh + (l - 1) * G_, LT_,
                                                last ? catb : xb64, last ? 160 : LT_, 0,
                                                nullptr, nullptr, 0);
  }

  // -------- contents path (T = 128, M = 65536) --------
  embed_sum_kernel<<<N_ * LC_, 160, 0, stream>>>(contents, emb, xb160, LC_, NC_);
  gemm_xg_kernel<5, 2><<<N_ * LC_ / 32, 128, 0, stream>>>(xb160, wih0b, ibih0, xg);
  gru_rec_kernel<<<N_ / 16, 128, 0, stream>>>(xg, iWhhB, ibhh0, LC_, xb64, LC_, 0,
                                              nullptr, nullptr, 0);
  for (int l = 1; l < 4; ++l) {
    gemm_xg_kernel<2, 4><<<N_ * LC_ / 64, 128, 0, stream>>>(xb64, iWihB + (l - 1) * WSZ,
                                                            ibih + (l - 1) * G_, xg);
    bool last = (l == 3);
    gru_rec_kernel<<<N_ / 16, 128, 0, stream>>>(xg, iWhhB + l * WSZ, ibhh + (l - 1) * G_, LC_,
                                                last ? catb : xb64, last ? 160 : LC_,
                                                last ? 32 : 0, nullptr, nullptr, 0);
  }

  // -------- comb path (T = 160, M = 81920) --------
  for (int l = 0; l < 4; ++l) {
    gemm_xg_kernel<2, 4><<<N_ * 160 / 64, 128, 0, stream>>>(l == 0 ? catb : xb64,
                                                            cWihB + l * WSZ, cbih + l * G_, xg);
    bool last = (l == 3);
    gru_rec_kernel<<<N_ / 16, 128, 0, stream>>>(xg, cWhhB + l * WSZ, cbhh + l * G_, 160,
                                                last ? nullptr : xb64, 160, 0,
                                                last ? outE : nullptr, outH, l);
  }
}
